// Bezier_10110353015098
// MI455X (gfx1250) — matrix operand order repaired and checked by reading, not compiled, not run on hardware
//
#include <hip/hip_runtime.h>
#include <hip/hip_bf16.h>

typedef __attribute__((ext_vector_type(16))) _Float16 v16h;
typedef __attribute__((ext_vector_type(8)))  _Float16 v8h;
typedef __attribute__((ext_vector_type(8)))  float    v8f;

#define RES   512
#define STEPS 128
#define NPTS  1024          // 8 curves * 128 steps
#define KDIM  1024
#define INV_2S2 5000.0f     // 1/(2*0.01^2)

// ---------------------------------------------------------------------------
// Kernel 1: De Casteljau cubic Bezier sampling.
// curves: (8,4,2) f32 row-major.  xy: [x[1024] | y[1024]] f32.
// Sample index k = curve*128 + step  (matches moveaxis+reshape in reference).
// ---------------------------------------------------------------------------
__global__ __launch_bounds__(256) void bezier_sample_kernel(
    const float* __restrict__ curves, float* __restrict__ xy) {
  int k = blockIdx.x * 256 + threadIdx.x;
  if (k >= NPTS) return;
  int c = k >> 7;          // curve index
  int s = k & 127;         // step index
  float t = (float)s * (1.0f / (STEPS - 1));
#pragma unroll
  for (int d = 0; d < 2; ++d) {
    float p0 = curves[c * 8 + 0 * 2 + d];
    float p1 = curves[c * 8 + 1 * 2 + d];
    float p2 = curves[c * 8 + 2 * 2 + d];
    float p3 = curves[c * 8 + 3 * 2 + d];
    float a01 = p0 + (p1 - p0) * t;
    float a12 = p1 + (p2 - p1) * t;
    float a23 = p2 + (p3 - p2) * t;
    float qa  = a01 + (a12 - a01) * t;
    float qb  = a12 + (a23 - a12) * t;
    float v   = qa + (qb - qa) * t;
    xy[d * NPTS + k] = v;
  }
}

// ---------------------------------------------------------------------------
// Kernel 2: Gaussian matrices in f16.
// E = [ Ex[512][1024] | Ey[512][1024] ], Ex[n][k] = exp(-(x_k - n/512)^2*5000)
// ---------------------------------------------------------------------------
__global__ __launch_bounds__(256) void gauss_kernel(
    const float* __restrict__ xy, _Float16* __restrict__ E) {
  int idx   = blockIdx.x * 256 + threadIdx.x;   // 0 .. 2*512*1024-1
  int which = idx >> 19;                        // 0 = Ex (x), 1 = Ey (y)
  int rem   = idx & ((RES * KDIM) - 1);
  int n     = rem >> 10;
  int k     = rem & (KDIM - 1);
  float v = xy[which * NPTS + k];
  float g = (float)n * (1.0f / (float)RES);
  float d = v - g;
  float e = __expf(-d * d * INV_2S2);
  E[idx] = (_Float16)e;
}

// ---------------------------------------------------------------------------
// Kernel 3: WMMA GEMM.  out[i*512+j] = sum_k Ey[i,k] * Ex[j,k]
//   A = Ey (M=512 rows over k), B[k][n] = Ex[n][k]  (contiguous in k!)
// 1024 tiles of 16x16; 8 waves/block, one tile per wave; K-loop of 32 WMMAs.
// Fragment layouts per CDNA5 ISA 7.12.2 (16-bit A 16x32, B 32x16, f32 C).
// ---------------------------------------------------------------------------
__global__ __launch_bounds__(256) void wmma_gemm_kernel(
    const _Float16* __restrict__ Ex, const _Float16* __restrict__ Ey,
    float* __restrict__ out) {
  int wave = threadIdx.x >> 5;
  int lane = threadIdx.x & 31;
  int tile = blockIdx.x * 8 + wave;    // 0..1023
  int tm = tile >> 5;                  // tile row    (0..31)
  int tn = tile & 31;                  // tile column (0..31)
  int m0 = tm * 16;
  int n0 = tn * 16;
  int lo16 = lane & 15;
  int hi   = lane >> 4;                // 0 for lanes 0-15, 1 for lanes 16-31

  const _Float16* arow = Ey + (size_t)(m0 + lo16) * KDIM;  // A row M=lane%16
  const _Float16* brow = Ex + (size_t)(n0 + lo16) * KDIM;  // B col N=lane%16

  v8f c = {};  // f32 accumulator, 8 VGPRs

#pragma unroll 4
  for (int k = 0; k < KDIM; k += 32) {
    // --- A fragment (16x32 f16): lanes 0-15 take K {0..7,16..23},
    //     lanes 16-31 take K {8..15,24..31}; two aligned 128-bit loads.
    v8h a_lo = *(const v8h*)(arow + k + 8 * hi);
    v8h a_hi = *(const v8h*)(arow + k + 16 + 8 * hi);
    // --- B fragment (32x16 f16): lanes 0-15 take K {0..7,16..23},
    //     lanes 16-31 take K {8..15,24..31}; B[k][n] = Ex[n][k] (row n of Ex).
    v8h b_lo = *(const v8h*)(brow + k + 8 * hi);
    v8h b_hi = *(const v8h*)(brow + k + 16 + 8 * hi);

    union { v16h v; v8h h[2]; } ua, ub;
    ua.h[0] = a_lo; ua.h[1] = a_hi;
    ub.h[0] = b_lo; ub.h[1] = b_hi;

    // (neg_a, A, neg_b, B, c_mod, C, reuse_a, reuse_b)
    c = __builtin_amdgcn_wmma_f32_16x16x32_f16(
        false, ua.v, false, ub.v, (short)0, c, false, false);
  }

  // --- C/D layout: VGPR r, lane L -> (M = r + 8*hi, N = L%16)
  float* ocol = out + n0 + lo16;
#pragma unroll
  for (int r = 0; r < 8; ++r) {
    ocol[(size_t)(m0 + r + 8 * hi) * RES] = c[r];
  }
}

// ---------------------------------------------------------------------------
extern "C" void kernel_launch(void* const* d_in, const int* in_sizes, int n_in,
                              void* d_out, int out_size, void* d_ws, size_t ws_size,
                              hipStream_t stream) {
  const float* curves = (const float*)d_in[0];   // 8*4*2 f32
  float* out = (float*)d_out;                    // 512*512 f32

  // Workspace layout: xy (2048 f32 = 8KB), then Ex|Ey (2*512*1024 f16 = 2MB)
  float*    xy = (float*)d_ws;
  _Float16* E  = (_Float16*)((char*)d_ws + 8192);
  _Float16* Ex = E;
  _Float16* Ey = E + (size_t)RES * KDIM;

  bezier_sample_kernel<<<(NPTS + 255) / 256, 256, 0, stream>>>(curves, xy);
  gauss_kernel<<<(2 * RES * KDIM) / 256, 256, 0, stream>>>(xy, E);
  wmma_gemm_kernel<<<128, 256, 0, stream>>>(Ex, Ey, out);
}
